// MultiHeadedSelfAttention_6743098655414
// MI455X (gfx1250) — matrix operand order repaired and checked by reading, not compiled, not run on hardware
//
#include <hip/hip_runtime.h>
#include <hip/hip_bf16.h>

// MHSA: B=4, S=2048, D=256, H=4, W=64, fp32 in/out.
// bf16 WMMA (v_wmma_f32_16x16x32_bf16) with f32 accumulation everywhere.
// Flash attention (transposed-score formulation) keeps the 256MB score tensor
// out of HBM; per-head K/V (512KB) streams through the 192MB L2 with 128x reuse.
// This revision: hardware bf16 converts (v_cvt_pk_bf16_f32 via
// __builtin_convertvector) replace software rounding, and the QKV store path
// is branchless (wave-uniform selects) to avoid EXEC-mask juggling.

static constexpr int Bb = 4;
static constexpr int Ss = 2048;
static constexpr int Dd = 256;
static constexpr int Hh = 4;
static constexpr int Ww = 64;

typedef __attribute__((ext_vector_type(16))) __bf16 bf16x16;
typedef __attribute__((ext_vector_type(2)))  __bf16 bf16x2;
typedef __attribute__((ext_vector_type(16))) unsigned short u16x16;
typedef __attribute__((ext_vector_type(2)))  float f32x2;
typedef __attribute__((ext_vector_type(8)))  float v8f;

union U16x16 { u16x16 v; unsigned int w[8]; };
union BF2U   { bf16x2 b; unsigned int u; unsigned short s[2]; };

// packed fp32x2 -> bf16x2 (hardware v_cvt_pk_bf16_f32)
static __device__ __forceinline__ unsigned int f2bf2(float x, float y) {
  f32x2 f = {x, y};
  BF2U r; r.b = __builtin_convertvector(f, bf16x2);
  return r.u;
}
static __device__ __forceinline__ unsigned short f2bf1(float x) {
  BF2U r; r.b = __builtin_convertvector((f32x2){x, 0.0f}, bf16x2);
  return r.s[0];
}
static __device__ __forceinline__ bf16x16 as_bf(u16x16 u) {
  union { u16x16 u; bf16x16 b; } x; x.u = u; return x.b;
}
// 16-bit A-matrix K-slot for (lane, vgpr-pair) per CDNA5 ISA 7.12.2:
// lanes 0-15: K = {0,1,...,7,16,...,23}; lanes 16-31: +8
static __device__ __forceinline__ int kslotA(int lane, int vg) {
  return ((lane >> 4) << 3) + ((vg < 4) ? (vg << 1) : (8 + (vg << 1)));
}
static __device__ __forceinline__ v8f wmma_bf16(u16x16 a, u16x16 b, v8f c) {
  return __builtin_amdgcn_wmma_f32_16x16x32_bf16(
      false, as_bf(a), false, as_bf(b), (short)0, c, false, false);
}

// ---------------- Kernel 1: fused QKV projection -----------------
// One wave per 16x16 output tile of one of {Q,K,V}. A = x rows (fp32->bf16),
// B = W columns. Q,K stored bf16 head-major (b,h,s,w); V stored TRANSPOSED
// (b,h,w,s) so flash V^T A-tiles load as contiguous bf16 pairs.
__global__ void __launch_bounds__(128)
qkv_kernel(const float* __restrict__ x,
           const float* __restrict__ Wq, const float* __restrict__ bq,
           const float* __restrict__ Wk, const float* __restrict__ bk,
           const float* __restrict__ Wv, const float* __restrict__ bv,
           unsigned short* __restrict__ Qh,
           unsigned short* __restrict__ Kh,
           unsigned short* __restrict__ Vt) {
  const int lane = threadIdx.x & 31;
  const int gw = blockIdx.x * (blockDim.x >> 5) + (threadIdx.x >> 5);
  const int MT = (Bb * Ss) / 16;  // 512
  const int NT = Dd / 16;         // 16
  const int mat = gw / (MT * NT);
  const int rem = gw - mat * (MT * NT);
  const int mtile = rem / NT;
  const int ntile = rem - mtile * NT;

  const float* Wm = (mat == 0) ? Wq : (mat == 1) ? Wk : Wv;
  const float* bm = (mat == 0) ? bq : (mat == 1) ? bk : bv;
  unsigned short* outp = (mat == 0) ? Qh : (mat == 1) ? Kh : Vt;

  const int mrow = mtile * 16 + (lane & 15);
  const int ncol = ntile * 16 + (lane & 15);
  const int kbaseB = lane & 16;     // A/B k order used here (contiguous): upper half-wave holds K+16
  const int hiA = (lane & 16) >> 1; // C/D-layout: upper half holds rows M+8

  v8f c = {};
#pragma unroll
  for (int kb = 0; kb < Dd; kb += 32) {
    U16x16 au, bu;
#pragma unroll
    for (int vg = 0; vg < 8; ++vg) {
      const float2 f = *(const float2*)(x + mrow * Dd + kb + kbaseB + (vg << 1));
      au.w[vg] = f2bf2(f.x, f.y);
      const int kc = kb + kbaseB + (vg << 1);
      bu.w[vg] = f2bf2(Wm[kc * Dd + ncol], Wm[(kc + 1) * Dd + ncol]);
    }
    c = wmma_bf16(au.v, bu.v, c);
  }
  const float bias = bm[ncol];
  const int hh = ncol >> 6, ww = ncol & 63;
  const bool isV = (mat == 2);
#pragma unroll
  for (int r = 0; r < 8; ++r) {
    const int grow = mtile * 16 + r + hiA;       // flat row over B*S
    const int b = grow >> 11;                    // /Ss
    const int s = grow & (Ss - 1);
    const int bh = b * Hh + hh;
    const int idx_qk = (bh * Ss + s) * Ww + ww;  // (b,h,s,w)
    const int idx_v  = (bh * Ww + ww) * Ss + s;  // (b,h,w,s) transposed
    outp[isV ? idx_v : idx_qk] = f2bf1(c[r] + bias);
  }
}

// ---------------- Kernel 2: flash attention ----------------------
// One wave owns a 16-query tile of one (b,h). Computes S^T = K*Q^T so the
// softmax key-dimension lies along VGPRs/half-waves, and O^T = V^T * P^T,
// all in registers. 8 WMMAs per 32-key block, 64 blocks per wave.
__global__ void __launch_bounds__(128)
flash_kernel(const unsigned short* __restrict__ Qh,
             const unsigned short* __restrict__ Kh,
             const unsigned short* __restrict__ Vt,
             unsigned short* __restrict__ Oc) {
  const int lane = threadIdx.x & 31;
  const int gw = blockIdx.x * (blockDim.x >> 5) + (threadIdx.x >> 5);
  const int QT = Ss / 16;          // 128 query tiles per (b,h)
  const int bh = gw / QT;
  const int qt = gw - bh * QT;
  const int b = bh / Hh, h = bh - b * Hh;
  const int qbase = qt * 16;
  const int q = qbase + (lane & 15);
  const int hiA = (lane & 16) >> 1;
  const bool lo = (lane < 16);

  // Q^T as two B-tiles (k=0..31, 32..63), register-resident for the whole loop
  U16x16 qB[2];
  const unsigned short* Qrow = Qh + (bh * Ss + q) * Ww;
#pragma unroll
  for (int cch = 0; cch < 2; ++cch)
#pragma unroll
    for (int vg = 0; vg < 8; ++vg)
      qB[cch].w[vg] =
          *(const unsigned int*)(Qrow + cch * 32 + kslotA(lane, vg));

  float m_run = -3.0e38f, l_run = 0.0f;
  v8f o[4] = {{}, {}, {}, {}};
  const unsigned short* Kbase = Kh + bh * Ss * Ww;
  const unsigned short* Vbase = Vt + bh * Ww * Ss;
  const float fi = (float)q;
  const float inv_s = 1.0f / (float)Ss;

  for (int kb = 0; kb < Ss; kb += 32) {
    // ---- scores S^T (2 tiles of 16 keys x 16 queries) ----
    v8f t0 = {}, t1 = {};
#pragma unroll
    for (int cch = 0; cch < 2; ++cch) {
      U16x16 a0, a1;
      const unsigned short* K0 = Kbase + (kb + (lane & 15)) * Ww + cch * 32;
      const unsigned short* K1 = K0 + 16 * Ww;
#pragma unroll
      for (int vg = 0; vg < 8; ++vg) {
        const int ks = kslotA(lane, vg);
        a0.w[vg] = *(const unsigned int*)(K0 + ks);
        a1.w[vg] = *(const unsigned int*)(K1 + ks);
      }
      t0 = wmma_bf16(a0.v, qB[cch].v, t0);
      t1 = wmma_bf16(a1.v, qB[cch].v, t1);
    }
    if (kb + 32 < Ss) {  // stream next K/V block toward the caches
      __builtin_prefetch(Kbase + (kb + 32 + (lane & 15)) * Ww, 0, 1);
      __builtin_prefetch(Vbase + (lane & 15) * Ss + kb + 32, 0, 1);
    }
    // ---- scale (1/sqrt(64)) and multiplicative distance-decay mask ----
#pragma unroll
    for (int r = 0; r < 8; ++r) {
      const float j0 = (float)(kb + r + hiA);
      const float j1 = j0 + 16.0f;
      t0[r] *= 0.125f * ((float)Ss - fabsf(fi - j0)) * inv_s;
      t1[r] *= 0.125f * ((float)Ss - fabsf(fi - j1)) * inv_s;
    }
    // ---- online softmax (key dim = VGPR index + half-wave) ----
    float bmx = -3.0e38f;
#pragma unroll
    for (int r = 0; r < 8; ++r) bmx = fmaxf(bmx, fmaxf(t0[r], t1[r]));
    bmx = fmaxf(bmx, __shfl_xor(bmx, 16, 32));
    const float m_new = fmaxf(m_run, bmx);
    const float alpha = __expf(m_run - m_new);
    float rs = 0.0f;
#pragma unroll
    for (int r = 0; r < 8; ++r) {
      t0[r] = __expf(t0[r] - m_new); rs += t0[r];
      t1[r] = __expf(t1[r] - m_new); rs += t1[r];
    }
    rs += __shfl_xor(rs, 16, 32);
    l_run = l_run * alpha + rs;
    m_run = m_new;
#pragma unroll
    for (int wt = 0; wt < 4; ++wt)
#pragma unroll
      for (int r = 0; r < 8; ++r) o[wt][r] *= alpha;

    // ---- repack P^T (C/D layout) into B-matrix layout: one half-wave swap ----
    float u0r[8], u1r[8];
#pragma unroll
    for (int r = 0; r < 8; ++r) {
      u0r[r] = __shfl_xor(t0[r], 16, 32);
      u1r[r] = __shfl_xor(t1[r], 16, 32);
    }
    U16x16 pB;
#pragma unroll
    for (int v = 0; v < 8; ++v) {
      const int j0 = 2 * v, j1 = 2 * v + 1;
      const float s0 = lo ? (j0 < 8 ? t0[j0] : u0r[j0 - 8])
                          : (j0 < 8 ? u1r[j0] : t1[j0 - 8]);
      const float s1 = lo ? (j1 < 8 ? t0[j1] : u0r[j1 - 8])
                          : (j1 < 8 ? u1r[j1] : t1[j1 - 8]);
      pB.w[v] = f2bf2(s0, s1);
    }
    // ---- O^T += V^T * P^T (4 w-tiles of 16) ----
#pragma unroll
    for (int wt = 0; wt < 4; ++wt) {
      U16x16 aV;
      const unsigned short* Vp = Vbase + (wt * 16 + (lane & 15)) * Ss + kb;
#pragma unroll
      for (int vg = 0; vg < 8; ++vg)
        aV.w[vg] = *(const unsigned int*)(Vp + (lane & 16) + (vg << 1));
      o[wt] = wmma_bf16(aV.v, pB.v, o[wt]);
    }
  }

  // ---- normalize and store O (row-major (B*S, D), bf16) ----
  const float inv = 1.0f / l_run;
  unsigned short* Orow = Oc + (b * Ss + q) * Dd + h * Ww;
#pragma unroll
  for (int wt = 0; wt < 4; ++wt)
#pragma unroll
    for (int r = 0; r < 8; r += 2)
      *(unsigned int*)(Orow + wt * 16 + hiA + r) =
          f2bf2(o[wt][r] * inv, o[wt][r + 1] * inv);
}

// ---------------- Kernel 3: output projection --------------------
__global__ void __launch_bounds__(128)
oproj_kernel(const unsigned short* __restrict__ Oc,
             const float* __restrict__ Wo, const float* __restrict__ bo,
             float* __restrict__ out) {
  const int lane = threadIdx.x & 31;
  const int gw = blockIdx.x * (blockDim.x >> 5) + (threadIdx.x >> 5);
  const int NT = Dd / 16;
  const int mtile = gw / NT;
  const int ntile = gw - mtile * NT;
  const int mrow = mtile * 16 + (lane & 15);
  const int ncol = ntile * 16 + (lane & 15);
  const int kbaseB = lane & 16;
  const int hiA = (lane & 16) >> 1;

  v8f c = {};
#pragma unroll
  for (int kb = 0; kb < Dd; kb += 32) {
    U16x16 au, bu;
#pragma unroll
    for (int vg = 0; vg < 8; ++vg) {
      au.w[vg] = *(const unsigned int*)(Oc + mrow * Dd + kb + kbaseB + (vg << 1));
      const int kc = kb + kbaseB + (vg << 1);
      bu.w[vg] = f2bf2(Wo[kc * Dd + ncol], Wo[(kc + 1) * Dd + ncol]);
    }
    c = wmma_bf16(au.v, bu.v, c);
  }
  const float bias = bo[ncol];
#pragma unroll
  for (int r = 0; r < 8; ++r) {
    const int grow = mtile * 16 + r + hiA;
    out[grow * Dd + ncol] = c[r] + bias;
  }
}

extern "C" void kernel_launch(void* const* d_in, const int* in_sizes, int n_in,
                              void* d_out, int out_size, void* d_ws, size_t ws_size,
                              hipStream_t stream) {
  (void)in_sizes; (void)n_in; (void)out_size; (void)ws_size;
  const float* x  = (const float*)d_in[0];
  const float* Wq = (const float*)d_in[1];
  const float* bq = (const float*)d_in[2];
  const float* Wk = (const float*)d_in[3];
  const float* bk = (const float*)d_in[4];
  const float* Wv = (const float*)d_in[5];
  const float* bv = (const float*)d_in[6];
  const float* Wo = (const float*)d_in[7];
  const float* bo = (const float*)d_in[8];

  const size_t elems = (size_t)Bb * Hh * Ss * Ww;  // 2M bf16 elems per tensor
  unsigned short* Qh = (unsigned short*)d_ws;
  unsigned short* Kh = Qh + elems;
  unsigned short* Vt = Kh + elems;
  unsigned short* Oc = Vt + elems;                 // (B*S, D) bf16

  // K1: 3 mats * 512 mtiles * 16 ntiles = 24576 waves, 4 waves/block
  qkv_kernel<<<6144, 128, 0, stream>>>(x, Wq, bq, Wk, bk, Wv, bv, Qh, Kh, Vt);
  // K2: 16 (b,h) * 128 query tiles = 2048 waves
  flash_kernel<<<512, 128, 0, stream>>>(Qh, Kh, Vt, Oc);
  // K3: 512 * 16 = 8192 waves
  oproj_kernel<<<2048, 128, 0, stream>>>(Oc, Wo, bo, (float*)d_out);
}
